// VAE_17952963297994
// MI455X (gfx1250) — compile-verified, matrix operand order checked by reading, not run on hardware
//
#include <hip/hip_runtime.h>

// ---------------- problem constants ----------------
#define B_   256
#define T_   256
#define I_   12
#define H_   512
#define Z_   64
#define TH_  1536   // 3*H

// ---------------- launch shape ---------------------
#define NWG  32     // must equal H_/16 (one 16-col block of H per workgroup)
#define BLK  256    // 8 wave32 waves per WG
#define NT   (NWG*BLK)

typedef __attribute__((ext_vector_type(16))) _Float16 v16h;
typedef __attribute__((ext_vector_type(8)))  float    v8f;

union FragU { v16h h; uint4 u[2]; };

struct P {
  const float *x, *eps;
  const float *w_ih_e, *b_ih_e, *b_hh_e;
  const float *w_ih_d, *b_ih_d, *b_hh_d;
  const float *fc1_w, *fc1_b, *fc11_w, *fc11_b, *fc12_w, *fc12_b;
  const float *fc2_w, *fc2_b, *fc3_w, *fc3_b, *fc4_w, *fc4_b;
  const _Float16 *whhE, *whhD;      // f16 copies of w_hh_e / w_hh_d, [1536][512] row-major
  const _Float16 *fc4h;             // f16 fc4_w padded to [16][512] (rows 12..15 = 0)
  float *h32a, *h32b;               // fp32 hidden, double buffered [256][512]
  _Float16 *h16a, *h16b;            // f16 hidden (WMMA A source), double buffered
  float *bufA, *bufB, *zbuf, *ubuf, *ybuf;
  float *out;                       // seq[B*T*I] | mu[B*Z] | logvar[B*Z]
  unsigned *cnt, *gen;
};

__device__ __forceinline__ float sigm(float x)   { return 1.f / (1.f + __expf(-x)); }
__device__ __forceinline__ float tanh_f(float x) { return 1.f - 2.f / (__expf(2.f * x) + 1.f); }

// device-wide sense-reversing barrier (32 co-resident WGs)
__device__ __forceinline__ void grid_barrier(unsigned* cnt, unsigned* gen) {
  __syncthreads();
  if (threadIdx.x == 0) {
    __threadfence();
    unsigned g = __hip_atomic_load(gen, __ATOMIC_RELAXED, __HIP_MEMORY_SCOPE_AGENT);
    unsigned a = __hip_atomic_fetch_add(cnt, 1u, __ATOMIC_ACQ_REL, __HIP_MEMORY_SCOPE_AGENT);
    if (a == gridDim.x - 1) {
      __hip_atomic_store(cnt, 0u, __ATOMIC_RELAXED, __HIP_MEMORY_SCOPE_AGENT);
      __hip_atomic_fetch_add(gen, 1u, __ATOMIC_RELEASE, __HIP_MEMORY_SCOPE_AGENT);
    } else {
      while (__hip_atomic_load(gen, __ATOMIC_ACQUIRE, __HIP_MEMORY_SCOPE_AGENT) == g)
        __builtin_amdgcn_s_sleep(2);
    }
    __threadfence();
  }
  __syncthreads();
}

// Stage the per-block constant weights for one GRU phase into LDS (ONCE per phase,
// not per step): w_hh slice (3 gates x 16 cols x 512 K, 48KB f16), plus the K=12
// w_ih slice and the bias slices used by the epilogue.
__device__ __forceinline__ void stage_phase(
    const _Float16* __restrict__ whh, const float* __restrict__ w_ih,
    const float* __restrict__ b_ih, const float* __restrict__ b_hh,
    _Float16* ldsW, float* ldsWih, float* ldsBih, float* ldsBhh)
{
  const int tid   = threadIdx.x;
  const int cBase = blockIdx.x * 16;
  const uint4* src = reinterpret_cast<const uint4*>(whh);
  uint4*       dst = reinterpret_cast<uint4*>(ldsW);
  for (int i = tid; i < 48 * 64; i += BLK) {   // 48 rows x 64 uint4/row
    int r = i >> 6, kc = i & 63;
    int g = r >> 4, cc = r & 15;
    dst[r * 64 + kc] = src[(g * H_ + cBase + cc) * (H_ / 8) + kc];
  }
  for (int i = tid; i < 48 * I_; i += BLK) {
    int r = i / I_, k = i % I_;
    int g = r >> 4, cc = r & 15;
    ldsWih[i] = w_ih[(g * H_ + cBase + cc) * I_ + k];
  }
  for (int i = tid; i < 48; i += BLK) {
    int g = i >> 4, cc = i & 15;
    ldsBih[i] = b_ih[g * H_ + cBase + cc];
    ldsBhh[i] = b_hh[g * H_ + cBase + cc];
  }
  __syncthreads();
}

// One fused GRU step:
//   gh = h @ w_hh^T  (WMMA f16->f32, B tiles from pre-staged LDS, shared by 8 waves)
//   gi = x_t @ w_ih^T (K=12, fp32 VALU, weights from LDS)
//   gate math + h update, writes h (fp32 + f16) to the "next" buffers.
__device__ __forceinline__ void gru_stage(
    const _Float16* __restrict__ h16p, const float* __restrict__ h32p,
    _Float16* __restrict__ h16n, float* __restrict__ h32n,
    const float* __restrict__ gi_src, int gi_stride,
    const _Float16* ldsW, const float* ldsWih,
    const float* ldsBih, const float* ldsBhh)
{
  const int tid   = threadIdx.x;
  const int lane  = tid & 31;
  const int wv    = tid >> 5;            // wave in WG: 0..7
  const int cBase = blockIdx.x * 16;     // this WG's 16-col block of H

  const int lr = lane & 15;
  const int hs = (lane & 16) ? 8 : 0;    // ISA 16-bit fragment K-half select
  const int mBase0 = wv * 32, mBase1 = wv * 32 + 16;
  const int rowA0 = mBase0 + lr, rowA1 = mBase1 + lr;

  v8f zero = {0.f,0.f,0.f,0.f,0.f,0.f,0.f,0.f};
  v8f acc[2][3];
  for (int q = 0; q < 2; ++q) for (int g = 0; g < 3; ++g) acc[q][g] = zero;

  const uint4* h4 = reinterpret_cast<const uint4*>(h16p);
  const uint4* l4 = reinterpret_cast<const uint4*>(ldsW);

  for (int kk = 0; kk < H_; kk += 32) {
    FragU a0, a1, br, bz, bn;
    int ao0 = (rowA0 * H_ + kk + hs) >> 3;  a0.u[0] = h4[ao0]; a0.u[1] = h4[ao0 + 2];
    int ao1 = (rowA1 * H_ + kk + hs) >> 3;  a1.u[0] = h4[ao1]; a1.u[1] = h4[ao1 + 2];
    int br0 = ((     lr) * H_ + kk + hs) >> 3;  br.u[0] = l4[br0]; br.u[1] = l4[br0 + 2];
    int bz0 = ((16 + lr) * H_ + kk + hs) >> 3;  bz.u[0] = l4[bz0]; bz.u[1] = l4[bz0 + 2];
    int bn0 = ((32 + lr) * H_ + kk + hs) >> 3;  bn.u[0] = l4[bn0]; bn.u[1] = l4[bn0 + 2];

    acc[0][0] = __builtin_amdgcn_wmma_f32_16x16x32_f16(false, a0.h, false, br.h, (short)0, acc[0][0], false, false);
    acc[0][1] = __builtin_amdgcn_wmma_f32_16x16x32_f16(false, a0.h, false, bz.h, (short)0, acc[0][1], false, false);
    acc[0][2] = __builtin_amdgcn_wmma_f32_16x16x32_f16(false, a0.h, false, bn.h, (short)0, acc[0][2], false, false);
    acc[1][0] = __builtin_amdgcn_wmma_f32_16x16x32_f16(false, a1.h, false, br.h, (short)0, acc[1][0], false, false);
    acc[1][1] = __builtin_amdgcn_wmma_f32_16x16x32_f16(false, a1.h, false, bz.h, (short)0, acc[1][1], false, false);
    acc[1][2] = __builtin_amdgcn_wmma_f32_16x16x32_f16(false, a1.h, false, bn.h, (short)0, acc[1][2], false, false);
  }

  // ---- epilogue: gi (K=12), biases, gate nonlinearity, h update ----
  const int col = cBase + lr;                 // C/D layout: N = lane%16 per lane
  float bihr = ldsBih[lr],      bhhr = ldsBhh[lr];
  float bihz = ldsBih[16 + lr], bhhz = ldsBhh[16 + lr];
  float bihn = ldsBih[32 + lr], bhhn = ldsBhh[32 + lr];
  float wr[I_], wz[I_], wn[I_];
  #pragma unroll
  for (int k = 0; k < I_; ++k) {
    wr[k] = ldsWih[lr * I_ + k];
    wz[k] = ldsWih[(16 + lr) * I_ + k];
    wn[k] = ldsWih[(32 + lr) * I_ + k];
  }
  #pragma unroll
  for (int q = 0; q < 2; ++q) {
    const int mB = q ? mBase1 : mBase0;
    #pragma unroll
    for (int j = 0; j < 8; ++j) {
      const int row = mB + j + ((lane & 16) ? 8 : 0);   // C/D layout: M = j (+8 hi lanes)
      const float* xr = gi_src + row * gi_stride;
      float gr = 0.f, gz = 0.f, gn = 0.f;
      #pragma unroll
      for (int k = 0; k < I_; ++k) {
        float xv = xr[k];
        gr = fmaf(xv, wr[k], gr); gz = fmaf(xv, wz[k], gz); gn = fmaf(xv, wn[k], gn);
      }
      float r  = sigm(acc[q][0][j] + bhhr + gr + bihr);
      float z  = sigm(acc[q][1][j] + bhhz + gz + bihz);
      float nn = tanh_f(gn + bihn + r * (acc[q][2][j] + bhhn));
      float hold = h32p[row * H_ + col];
      float hnew = (1.f - z) * nn + z * hold;
      h32n[row * H_ + col] = hnew;
      h16n[row * H_ + col] = (_Float16)hnew;
    }
  }
}

// Decoder output head: y_t = h_t @ fc4^T + b via WMMA on zero-padded f16 fc4 [16][512].
// 16 waves, one 16x16 output tile each (only the 12 valid columns are written).
__device__ __forceinline__ void y_stage(
    const _Float16* __restrict__ h16, const _Float16* __restrict__ fc4h,
    const float* __restrict__ fc4_b, float* __restrict__ ybuf,
    float* __restrict__ out, int t)
{
  const int waveId = (blockIdx.x * BLK + threadIdx.x) >> 5;
  if (waveId >= 16) return;                 // wave-uniform branch, EXEC stays full
  const int lane = threadIdx.x & 31;
  const int lr   = lane & 15;
  const int hs   = (lane & 16) ? 8 : 0;
  const int mBase = waveId * 16;
  const int rowA  = mBase + lr;

  v8f acc = {0.f,0.f,0.f,0.f,0.f,0.f,0.f,0.f};
  const uint4* h4 = reinterpret_cast<const uint4*>(h16);
  const uint4* w4 = reinterpret_cast<const uint4*>(fc4h);

  for (int kk = 0; kk < H_; kk += 32) {
    FragU a, b;
    int ao = (rowA * H_ + kk + hs) >> 3;  a.u[0] = h4[ao]; a.u[1] = h4[ao + 2];
    int bo = (lr   * H_ + kk + hs) >> 3;  b.u[0] = w4[bo]; b.u[1] = w4[bo + 2];
    acc = __builtin_amdgcn_wmma_f32_16x16x32_f16(false, a.h, false, b.h, (short)0, acc, false, false);
  }

  if (lr < I_) {
    const float bias = fc4_b[lr];
    #pragma unroll
    for (int j = 0; j < 8; ++j) {
      const int row = mBase + j + ((lane & 16) ? 8 : 0);
      float s = acc[j] + bias;
      ybuf[row * I_ + lr] = s;
      out[(row * T_ + t) * I_ + lr] = s;
    }
  }
}

__global__ void __launch_bounds__(BLK, 1) vae_persist(P p) {
  __shared__ _Float16 ldsW[48 * H_];     // 48 KB staged w_hh slice (per phase)
  __shared__ float    ldsWih[48 * I_];   // staged w_ih slice
  __shared__ float    ldsBih[48], ldsBhh[48];
  const int gtid = blockIdx.x * BLK + threadIdx.x;

  // ---- phase 0: zero initial hidden state ----
  for (int i = gtid; i < B_ * H_; i += NT) { p.h32a[i] = 0.f; p.h16a[i] = (_Float16)0.f; }
  grid_barrier(p.cnt, p.gen);

  // ---- encoder GRU over T steps (weights staged ONCE, h double-buffered) ----
  {
    stage_phase(p.whhE, p.w_ih_e, p.b_ih_e, p.b_hh_e, ldsW, ldsWih, ldsBih, ldsBhh);
    const float*    h32p = p.h32a; float*    h32n = p.h32b;
    const _Float16* h16p = p.h16a; _Float16* h16n = p.h16b;
    for (int t = 0; t < T_; ++t) {
      gru_stage(h16p, h32p, h16n, h32n, p.x + t * I_, T_ * I_,
                ldsW, ldsWih, ldsBih, ldsBhh);
      grid_barrier(p.cnt, p.gen);
      const float* t32 = h32p; h32p = h32n; h32n = (float*)t32;
      const _Float16* t16 = h16p; h16p = h16n; h16n = (_Float16*)t16;
    }
  } // final h_enc (fp32) is in p.h32a (even number of swaps)

  // ---- latent head (small fp32 dense layers, distributed over the grid) ----
  for (int i = gtid; i < B_ * H_; i += NT) {          // v = lrelu(h_enc)
    float v = p.h32a[i]; p.bufA[i] = fmaxf(v, 0.01f * v);
  }
  grid_barrier(p.cnt, p.gen);

  for (int i = gtid; i < B_ * 256; i += NT) {         // v1 = lrelu(v @ fc1^T + b)
    int m = i >> 8, n = i & 255;
    const float* a = p.bufA + m * H_; const float* w = p.fc1_w + n * H_;
    float s = p.fc1_b[n];
    for (int k = 0; k < H_; ++k) s = fmaf(a[k], w[k], s);
    p.bufB[i] = fmaxf(s, 0.01f * s);
  }
  grid_barrier(p.cnt, p.gen);

  {                                                   // mu, logvar, z = mu + exp(lv/2)*eps
    float* outmu = p.out + B_ * T_ * I_;
    float* outlv = outmu + B_ * Z_;
    for (int i = gtid; i < B_ * Z_; i += NT) {
      int m = i >> 6, n = i & 63;
      const float* a  = p.bufB + m * 256;
      const float* wm = p.fc11_w + n * 256; const float* wl = p.fc12_w + n * 256;
      float sm = p.fc11_b[n], sl = p.fc12_b[n];
      for (int k = 0; k < 256; ++k) { float av = a[k]; sm = fmaf(av, wm[k], sm); sl = fmaf(av, wl[k], sl); }
      outmu[i] = sm; outlv[i] = sl;
      p.zbuf[i] = fmaf(__expf(0.5f * sl), p.eps[i], sm);
    }
  }
  grid_barrier(p.cnt, p.gen);

  for (int i = gtid; i < B_ * 256; i += NT) {         // u = lrelu(z @ fc2^T + b)
    int m = i >> 8, n = i & 255;
    const float* a = p.zbuf + m * Z_; const float* w = p.fc2_w + n * Z_;
    float s = p.fc2_b[n];
    for (int k = 0; k < Z_; ++k) s = fmaf(a[k], w[k], s);
    p.ubuf[i] = fmaxf(s, 0.01f * s);
  }
  grid_barrier(p.cnt, p.gen);

  for (int i = gtid; i < B_ * H_; i += NT) {          // h_dec = u @ fc3^T + b
    int m = i >> 9, n = i & 511;
    const float* a = p.ubuf + m * 256; const float* w = p.fc3_w + n * 256;
    float s = p.fc3_b[n];
    for (int k = 0; k < 256; ++k) s = fmaf(a[k], w[k], s);
    p.h32a[i] = s; p.h16a[i] = (_Float16)s;
  }
  for (int i = gtid; i < B_ * I_; i += NT) p.ybuf[i] = 0.f;   // y0 = 0
  grid_barrier(p.cnt, p.gen);

  // ---- autoregressive decoder ----
  {
    stage_phase(p.whhD, p.w_ih_d, p.b_ih_d, p.b_hh_d, ldsW, ldsWih, ldsBih, ldsBhh);
    const float*    h32p = p.h32a; float*    h32n = p.h32b;
    const _Float16* h16p = p.h16a; _Float16* h16n = p.h16b;
    for (int t = 0; t < T_; ++t) {
      gru_stage(h16p, h32p, h16n, h32n, p.ybuf, I_,
                ldsW, ldsWih, ldsBih, ldsBhh);
      grid_barrier(p.cnt, p.gen);
      y_stage(h16n, p.fc4h, p.fc4_b, p.ybuf, p.out, t);   // y_t = h_t @ fc4^T + b (WMMA)
      grid_barrier(p.cnt, p.gen);
      const float* t32 = h32p; h32p = h32n; h32n = (float*)t32;
      const _Float16* t16 = h16p; h16p = h16n; h16n = (_Float16*)t16;
    }
  }
}

// convert recurrent weights (and the padded fc4 head) to f16 + reset barrier state
__global__ void prep_kernel(const float* we, const float* wd, const float* fc4w,
                            _Float16* oe, _Float16* od, _Float16* o4,
                            unsigned* cnt, unsigned* gen) {
  const int n  = TH_ * H_;
  const int gt = blockIdx.x * blockDim.x + threadIdx.x;
  const int st = gridDim.x * blockDim.x;
  for (int i = gt; i < n; i += st) { oe[i] = (_Float16)we[i]; od[i] = (_Float16)wd[i]; }
  for (int i = gt; i < 16 * H_; i += st) {
    int row = i >> 9;   // H_ == 512
    o4[i] = (row < I_) ? (_Float16)fc4w[i] : (_Float16)0.f;
  }
  if (gt == 0) { *cnt = 0u; *gen = 0u; }
}

extern "C" void kernel_launch(void* const* d_in, const int* in_sizes, int n_in,
                              void* d_out, int out_size, void* d_ws, size_t ws_size,
                              hipStream_t stream) {
  (void)in_sizes; (void)n_in; (void)out_size; (void)ws_size;

  char*  ws = (char*)d_ws;
  size_t o  = 0;
  auto alloc = [&](size_t bytes) -> void* {
    void* r = ws + o;
    o = (o + bytes + 255) & ~size_t(255);
    return r;
  };

  P p;
  p.x      = (const float*)d_in[0];
  p.eps    = (const float*)d_in[1];
  p.w_ih_e = (const float*)d_in[2];
  const float* w_hh_e = (const float*)d_in[3];
  p.b_ih_e = (const float*)d_in[4];
  p.b_hh_e = (const float*)d_in[5];
  p.w_ih_d = (const float*)d_in[6];
  const float* w_hh_d = (const float*)d_in[7];
  p.b_ih_d = (const float*)d_in[8];
  p.b_hh_d = (const float*)d_in[9];
  p.fc1_w  = (const float*)d_in[10];  p.fc1_b  = (const float*)d_in[11];
  p.fc11_w = (const float*)d_in[12];  p.fc11_b = (const float*)d_in[13];
  p.fc12_w = (const float*)d_in[14];  p.fc12_b = (const float*)d_in[15];
  p.fc2_w  = (const float*)d_in[16];  p.fc2_b  = (const float*)d_in[17];
  p.fc3_w  = (const float*)d_in[18];  p.fc3_b  = (const float*)d_in[19];
  p.fc4_w  = (const float*)d_in[20];  p.fc4_b  = (const float*)d_in[21];

  _Float16* whhE = (_Float16*)alloc((size_t)TH_ * H_ * 2);
  _Float16* whhD = (_Float16*)alloc((size_t)TH_ * H_ * 2);
  _Float16* fc4h = (_Float16*)alloc((size_t)16 * H_ * 2);
  p.whhE = whhE; p.whhD = whhD; p.fc4h = fc4h;
  p.h32a = (float*)alloc((size_t)B_ * H_ * 4);
  p.h32b = (float*)alloc((size_t)B_ * H_ * 4);
  p.h16a = (_Float16*)alloc((size_t)B_ * H_ * 2);
  p.h16b = (_Float16*)alloc((size_t)B_ * H_ * 2);
  p.bufA = (float*)alloc((size_t)B_ * H_ * 4);
  p.bufB = (float*)alloc((size_t)B_ * 256 * 4);
  p.zbuf = (float*)alloc((size_t)B_ * Z_ * 4);
  p.ubuf = (float*)alloc((size_t)B_ * 256 * 4);
  p.ybuf = (float*)alloc((size_t)B_ * I_ * 4);
  p.cnt  = (unsigned*)alloc(64);
  p.gen  = (unsigned*)alloc(64);
  p.out  = (float*)d_out;

  prep_kernel<<<256, 256, 0, stream>>>(w_hh_e, w_hh_d, p.fc4_w, whhE, whhD, fc4h, p.cnt, p.gen);
  vae_persist<<<NWG, BLK, 0, stream>>>(p);
}
